// Model_87041807221069
// MI455X (gfx1250) — matrix operand order repaired and checked by reading, not compiled, not run on hardware
//
#include <hip/hip_runtime.h>

typedef __attribute__((ext_vector_type(16))) _Float16 v16h;
typedef __attribute__((ext_vector_type(8)))  _Float16 v8h;
typedef __attribute__((ext_vector_type(8)))  float    v8f;
typedef __attribute__((ext_vector_type(4)))  int      v4i;

struct __align__(8) h4 { _Float16 x, y, z, w; };

static __device__ inline v16h load16(const _Float16* p0, const _Float16* p1) {
    v8h lo = *(const v8h*)p0;
    v8h hi = *(const v8h*)p1;
    return __builtin_shufflevector(lo, hi, 0,1,2,3,4,5,6,7,8,9,10,11,12,13,14,15);
}

static __device__ inline float relu(float v) { return v > 0.0f ? v : 0.0f; }

// 16B async global->LDS copy (CDNA5 GLOBAL_LOAD_ASYNC_TO_LDS_B128, ASYNCcnt)
static __device__ inline void async_cp16(_Float16* ldst, const _Float16* gsrc) {
#if __has_builtin(__builtin_amdgcn_global_load_async_to_lds_b128)
    __builtin_amdgcn_global_load_async_to_lds_b128(
        (__attribute__((address_space(1))) v4i*)(v4i*)(void*)const_cast<_Float16*>(gsrc),
        (__attribute__((address_space(3))) v4i*)(v4i*)(void*)ldst,
        0, 0);
#else
    *(uint4*)ldst = *(const uint4*)gsrc;
#endif
}

static __device__ inline void wait_async0() {
#if __has_builtin(__builtin_amdgcn_s_wait_asynccnt)
    __builtin_amdgcn_s_wait_asynccnt(0);
#else
    asm volatile("s_wait_asynccnt 0" ::: "memory");
#endif
}

// ---------------------------------------------------------------------------
// utility kernels
// ---------------------------------------------------------------------------
__global__ void zero_kernel(float* __restrict__ p, size_t n) {
    size_t i = (size_t)blockIdx.x * blockDim.x + threadIdx.x;
    size_t stride = (size_t)gridDim.x * blockDim.x;
    for (; i < n; i += stride) p[i] = 0.0f;
}

__global__ void count_kernel(const int* __restrict__ dst, float* __restrict__ cnt, int E) {
    int i = blockIdx.x * blockDim.x + threadIdx.x;
    if (i < E) atomicAdd(&cnt[dst[i]], 1.0f);
}

__global__ void cvt16_kernel(const float* __restrict__ in, _Float16* __restrict__ out, size_t n4) {
    size_t i = (size_t)blockIdx.x * blockDim.x + threadIdx.x;
    size_t stride = (size_t)gridDim.x * blockDim.x;
    for (; i < n4; i += stride) {
        float4 v = ((const float4*)in)[i];
        h4 p = { (_Float16)v.x, (_Float16)v.y, (_Float16)v.z, (_Float16)v.w };
        ((h4*)out)[i] = p;
    }
}

// ---------------------------------------------------------------------------
// edge message: m = relu([h_src || e] @ W + b); atomic scatter-add into s[dst]
// h_src is f16 [*, KSRC]; edge features are f32 [E, KEDGE]. Out width = 128.
// ---------------------------------------------------------------------------
template<int KSRC, int KEDGE>
__global__ __launch_bounds__(256) void edge_msg_kernel(
    const _Float16* __restrict__ hsrc, const float* __restrict__ ef,
    const int* __restrict__ src, const int* __restrict__ dst,
    const float* __restrict__ W, const float* __restrict__ bias,
    float* __restrict__ sOut, int E, int numTiles)
{
    constexpr int K  = KSRC + KEDGE;     // multiple of 32
    constexpr int AS = K + 8;            // padded row stride (halves), 16B multiple
    constexpr int C1 = KSRC / 8;         // 16B f16 chunks per row (async)
    constexpr int C2 = KEDGE / 4;        // float4 chunks per row (convert)

    __shared__ __align__(16) _Float16 Alds[128 * AS];
    __shared__ __align__(16) _Float16 Wlds[128 * AS];
    __shared__ int sIdx[128];
    __shared__ int dIdx[128];

    const int tid  = threadIdx.x;
    const int lane = tid & 31;
    const int wv   = tid >> 5;
    const int l15  = lane & 15;
    const int hi   = lane >> 4;

    // stage transposed f16 weights once per block: Wlds[n][k] = W[k][n]
    for (int idx = tid; idx < 128 * K; idx += 256) {
        int n = idx & 127, k = idx >> 7;
        Wlds[n * AS + k] = (_Float16)W[k * 128 + n];
    }
    float breg[8];
#pragma unroll
    for (int t = 0; t < 8; ++t) breg[t] = bias[t * 16 + l15];

    for (int tile = blockIdx.x; tile < numTiles; tile += gridDim.x) {
        const int e0 = tile * 128;
        __syncthreads();                       // prior tile's LDS reads done
        if (tid < 128) {
            int e = e0 + tid;
            sIdx[tid] = (e < E) ? src[e] : 0;  // OOB rows alias row 0; discarded later
            dIdx[tid] = (e < E) ? dst[e] : 0;
        }
        __syncthreads();

        // gather h_src (f16) via async global->LDS
        for (int idx = tid; idx < 128 * C1; idx += 256) {
            int le = idx / C1, c = idx % C1;
            async_cp16(&Alds[le * AS + c * 8],
                       hsrc + (size_t)sIdx[le] * KSRC + c * 8);
        }
        // gather edge features (f32 -> f16 in flight)
        for (int idx = tid; idx < 128 * C2; idx += 256) {
            int le = idx / C2, c = idx % C2;
            int e  = e0 + le;
            float4 v = make_float4(0.f, 0.f, 0.f, 0.f);
            if (e < E) v = ((const float4*)ef)[(size_t)e * C2 + c];
            h4 p = { (_Float16)v.x, (_Float16)v.y, (_Float16)v.z, (_Float16)v.w };
            *(h4*)&Alds[le * AS + KSRC + c * 4] = p;
        }
        wait_async0();
        __syncthreads();

        // 16 edges x 128 outputs per wave via WMMA
        const _Float16* arow = &Alds[(wv * 16 + l15) * AS];
        v8f acc[8];
#pragma unroll
        for (int t = 0; t < 8; ++t) {
            v8f a;
#pragma unroll
            for (int r = 0; r < 8; ++r) a[r] = breg[t];
            acc[t] = a;
        }
#pragma unroll
        for (int kk = 0; kk < K / 32; ++kk) {
            const int kb = kk * 32;
            v16h afrag = load16(arow + kb + hi * 8, arow + kb + hi * 8 + 16);
#pragma unroll
            for (int t = 0; t < 8; ++t) {
                const _Float16* wrow = &Wlds[(t * 16 + l15) * AS + kb + hi * 8];
                v16h bfrag = load16(wrow, wrow + 16);
                acc[t] = __builtin_amdgcn_wmma_f32_16x16x32_f16(
                    false, afrag, false, bfrag, (short)0, acc[t], false, false);
            }
        }

        // relu + atomic scatter into segment sums
#pragma unroll
        for (int t = 0; t < 8; ++t) {
#pragma unroll
            for (int r = 0; r < 8; ++r) {
                int M  = hi * 8 + r;
                int le = wv * 16 + M;
                int e  = e0 + le;
                if (e < E) {
                    float v = relu(acc[t][r]);
                    atomicAdd(&sOut[(size_t)dIdx[le] * 128 + t * 16 + l15], v);
                }
            }
        }
    }
}

// ---------------------------------------------------------------------------
// node apply: h = relu([h_prev || s/max(cnt,1)] @ W + b), out width 128 (f16)
// ---------------------------------------------------------------------------
template<int K1>
__global__ __launch_bounds__(256) void node_apply_kernel(
    const _Float16* __restrict__ hprev, const float* __restrict__ sIn,
    const float* __restrict__ cnt,
    const float* __restrict__ W, const float* __restrict__ bias,
    _Float16* __restrict__ hOut, int Nn, int numTiles)
{
    constexpr int K  = K1 + 128;
    constexpr int AS = K + 8;
    constexpr int C1 = K1 / 8;

    __shared__ __align__(16) _Float16 Alds[128 * AS];
    __shared__ __align__(16) _Float16 Wlds[128 * AS];
    __shared__ float rcpL[128];

    const int tid  = threadIdx.x;
    const int lane = tid & 31;
    const int wv   = tid >> 5;
    const int l15  = lane & 15;
    const int hi   = lane >> 4;

    for (int idx = tid; idx < 128 * K; idx += 256) {
        int n = idx & 127, k = idx >> 7;
        Wlds[n * AS + k] = (_Float16)W[k * 128 + n];
    }
    float breg[8];
#pragma unroll
    for (int t = 0; t < 8; ++t) breg[t] = bias[t * 16 + l15];

    for (int tile = blockIdx.x; tile < numTiles; tile += gridDim.x) {
        const int n0 = tile * 128;
        __syncthreads();
        if (tid < 128) {
            int node = n0 + tid;
            rcpL[tid] = (node < Nn) ? 1.0f / fmaxf(cnt[node], 1.0f) : 0.0f;
        }
        __syncthreads();

        // h_prev rows (f16, block-contiguous) via async global->LDS
        for (int idx = tid; idx < 128 * C1; idx += 256) {
            int ln = idx / C1, c = idx % C1;
            int node = n0 + ln;
            if (node >= Nn) node = 0;
            async_cp16(&Alds[ln * AS + c * 8],
                       hprev + (size_t)node * K1 + c * 8);
        }
        // neighbor mean (f32 sums scaled, converted in flight)
        for (int idx = tid; idx < 128 * 32; idx += 256) {
            int ln = idx >> 5, c = idx & 31;
            int node = n0 + ln;
            float4 v = make_float4(0.f, 0.f, 0.f, 0.f);
            if (node < Nn) {
                v = ((const float4*)sIn)[(size_t)node * 32 + c];
                float rc = rcpL[ln];
                v.x *= rc; v.y *= rc; v.z *= rc; v.w *= rc;
            }
            h4 p = { (_Float16)v.x, (_Float16)v.y, (_Float16)v.z, (_Float16)v.w };
            *(h4*)&Alds[ln * AS + K1 + c * 4] = p;
        }
        wait_async0();
        __syncthreads();

        const _Float16* arow = &Alds[(wv * 16 + l15) * AS];
        v8f acc[8];
#pragma unroll
        for (int t = 0; t < 8; ++t) {
            v8f a;
#pragma unroll
            for (int r = 0; r < 8; ++r) a[r] = breg[t];
            acc[t] = a;
        }
#pragma unroll
        for (int kk = 0; kk < K / 32; ++kk) {
            const int kb = kk * 32;
            v16h afrag = load16(arow + kb + hi * 8, arow + kb + hi * 8 + 16);
#pragma unroll
            for (int t = 0; t < 8; ++t) {
                const _Float16* wrow = &Wlds[(t * 16 + l15) * AS + kb + hi * 8];
                v16h bfrag = load16(wrow, wrow + 16);
                acc[t] = __builtin_amdgcn_wmma_f32_16x16x32_f16(
                    false, afrag, false, bfrag, (short)0, acc[t], false, false);
            }
        }
#pragma unroll
        for (int t = 0; t < 8; ++t) {
#pragma unroll
            for (int r = 0; r < 8; ++r) {
                int M    = hi * 8 + r;
                int node = n0 + wv * 16 + M;
                if (node < Nn)
                    hOut[(size_t)node * 128 + t * 16 + l15] = (_Float16)relu(acc[t][r]);
            }
        }
    }
}

// ---------------------------------------------------------------------------
// predictor: out = relu([h_u || h_v] @ W1 + b1) @ W2 + b2   (W2: 128x2)
// ---------------------------------------------------------------------------
__global__ __launch_bounds__(256) void edge_pred_kernel(
    const _Float16* __restrict__ h2, const int* __restrict__ src,
    const int* __restrict__ dst,
    const float* __restrict__ W1, const float* __restrict__ b1,
    const float* __restrict__ W2, const float* __restrict__ b2,
    float* __restrict__ out, int E, int numTiles)
{
    constexpr int K  = 256;
    constexpr int AS = K + 8;
    constexpr int C1 = 32;               // 16B f16 chunks per 256-wide row

    __shared__ __align__(16) _Float16 Alds[128 * AS];
    __shared__ __align__(16) _Float16 Wlds[128 * AS];
    __shared__ int   sIdx[128];
    __shared__ int   dIdx[128];
    __shared__ float w2l[256];

    const int tid  = threadIdx.x;
    const int lane = tid & 31;
    const int wv   = tid >> 5;
    const int l15  = lane & 15;
    const int hi   = lane >> 4;

    for (int idx = tid; idx < 128 * K; idx += 256) {
        int n = idx & 127, k = idx >> 7;
        Wlds[n * AS + k] = (_Float16)W1[k * 128 + n];
    }
    if (tid < 256) w2l[tid] = W2[tid];
    float breg[8];
#pragma unroll
    for (int t = 0; t < 8; ++t) breg[t] = b1[t * 16 + l15];
    const float b2v0 = b2[0], b2v1 = b2[1];

    for (int tile = blockIdx.x; tile < numTiles; tile += gridDim.x) {
        const int e0 = tile * 128;
        __syncthreads();
        if (tid < 128) {
            int e = e0 + tid;
            sIdx[tid] = (e < E) ? src[e] : 0;
            dIdx[tid] = (e < E) ? dst[e] : 0;
        }
        __syncthreads();

        // gather [h_u || h_v] (f16) fully via async global->LDS
        for (int idx = tid; idx < 128 * C1; idx += 256) {
            int le = idx / C1, c = idx % C1;
            const _Float16* g = (c < 16)
                ? h2 + (size_t)sIdx[le] * 128 + c * 8
                : h2 + (size_t)dIdx[le] * 128 + (c - 16) * 8;
            async_cp16(&Alds[le * AS + c * 8], g);
        }
        wait_async0();
        __syncthreads();

        const _Float16* arow = &Alds[(wv * 16 + l15) * AS];
        v8f acc[8];
#pragma unroll
        for (int t = 0; t < 8; ++t) {
            v8f a;
#pragma unroll
            for (int r = 0; r < 8; ++r) a[r] = breg[t];
            acc[t] = a;
        }
#pragma unroll
        for (int kk = 0; kk < K / 32; ++kk) {
            const int kb = kk * 32;
            v16h afrag = load16(arow + kb + hi * 8, arow + kb + hi * 8 + 16);
#pragma unroll
            for (int t = 0; t < 8; ++t) {
                const _Float16* wrow = &Wlds[(t * 16 + l15) * AS + kb + hi * 8];
                v16h bfrag = load16(wrow, wrow + 16);
                acc[t] = __builtin_amdgcn_wmma_f32_16x16x32_f16(
                    false, afrag, false, bfrag, (short)0, acc[t], false, false);
            }
        }

        // second GEMM (128 -> 2): per-lane partials + xor-shuffle reduce over 16 lanes
        float s0[8], s1[8];
#pragma unroll
        for (int r = 0; r < 8; ++r) { s0[r] = 0.f; s1[r] = 0.f; }
#pragma unroll
        for (int t = 0; t < 8; ++t) {
            int n = t * 16 + l15;
            float w20 = w2l[n * 2 + 0], w21 = w2l[n * 2 + 1];
#pragma unroll
            for (int r = 0; r < 8; ++r) {
                float v = relu(acc[t][r]);
                s0[r] += v * w20;
                s1[r] += v * w21;
            }
        }
#pragma unroll
        for (int off = 1; off < 16; off <<= 1) {
#pragma unroll
            for (int r = 0; r < 8; ++r) {
                s0[r] += __shfl_xor(s0[r], off, 32);
                s1[r] += __shfl_xor(s1[r], off, 32);
            }
        }
        if (l15 < 8) {
            int M = hi * 8 + l15;
            int e = e0 + wv * 16 + M;
            if (e < E) {
                out[(size_t)e * 2 + 0] = s0[l15] + b2v0;
                out[(size_t)e * 2 + 1] = s1[l15] + b2v1;
            }
        }
    }
}

// ---------------------------------------------------------------------------
extern "C" void kernel_launch(void* const* d_in, const int* in_sizes, int n_in,
                              void* d_out, int out_size, void* d_ws, size_t ws_size,
                              hipStream_t stream) {
    const float* nfeats = (const float*)d_in[0];
    const float* efeats = (const float*)d_in[1];
    const int*   src    = (const int*)d_in[2];
    const int*   dst    = (const int*)d_in[3];
    const float* Wm1 = (const float*)d_in[4];
    const float* bm1 = (const float*)d_in[5];
    const float* Wa1 = (const float*)d_in[6];
    const float* ba1 = (const float*)d_in[7];
    const float* Wm2 = (const float*)d_in[8];
    const float* bm2 = (const float*)d_in[9];
    const float* Wa2 = (const float*)d_in[10];
    const float* ba2 = (const float*)d_in[11];
    const float* W1  = (const float*)d_in[12];
    const float* b1  = (const float*)d_in[13];
    const float* W2  = (const float*)d_in[14];
    const float* b2  = (const float*)d_in[15];
    float* out = (float*)d_out;

    const int Nn = in_sizes[0] / 64;   // 50000
    const int E  = in_sizes[2];        // 1600000

    auto align256 = [](size_t x) { return (x + 255) & ~(size_t)255; };
    char* ws = (char*)d_ws;
    size_t o = 0;
    float*     sBuf   = (float*)(ws + o);     o = align256(o + (size_t)Nn * 128 * 4);
    float*     cntBuf = (float*)(ws + o);     o = align256(o + (size_t)Nn * 4);
    _Float16*  nf16   = (_Float16*)(ws + o);  o = align256(o + (size_t)Nn * 64 * 2);
    _Float16*  h1     = (_Float16*)(ws + o);  o = align256(o + (size_t)Nn * 128 * 2);
    _Float16*  h2     = (_Float16*)(ws + o);

    const int edgeTiles = (E + 127) / 128;
    const int nodeTiles = (Nn + 127) / 128;
    const int edgeGrid  = edgeTiles < 2048 ? edgeTiles : 2048;

    // prep: zero accumulators, degree counts, f16 node features
    zero_kernel<<<2048, 256, 0, stream>>>(sBuf, (size_t)Nn * 128);
    zero_kernel<<<256, 256, 0, stream>>>(cntBuf, (size_t)Nn);
    count_kernel<<<(E + 255) / 256, 256, 0, stream>>>(dst, cntBuf, E);
    cvt16_kernel<<<512, 256, 0, stream>>>(nfeats, nf16, (size_t)Nn * 16);

    // layer 1
    edge_msg_kernel<64, 32><<<edgeGrid, 256, 0, stream>>>(
        nf16, efeats, src, dst, Wm1, bm1, sBuf, E, edgeTiles);
    node_apply_kernel<64><<<nodeTiles, 256, 0, stream>>>(
        nf16, sBuf, cntBuf, Wa1, ba1, h1, Nn, nodeTiles);

    // layer 2
    zero_kernel<<<2048, 256, 0, stream>>>(sBuf, (size_t)Nn * 128);
    edge_msg_kernel<128, 32><<<edgeGrid, 256, 0, stream>>>(
        h1, efeats, src, dst, Wm2, bm2, sBuf, E, edgeTiles);
    node_apply_kernel<128><<<nodeTiles, 256, 0, stream>>>(
        h1, sBuf, cntBuf, Wa2, ba2, h2, Nn, nodeTiles);

    // edge predictor
    edge_pred_kernel<<<edgeGrid, 256, 0, stream>>>(
        h2, src, dst, W1, b1, W2, b2, out, E, edgeTiles);
}
